// TNOBlock_70162585747981
// MI455X (gfx1250) — matrix operand order repaired and checked by reading, not compiled, not run on hardware
//
#include <hip/hip_runtime.h>
#include <hip/hip_bf16.h>

// ---------------------------------------------------------------------------
// TNO block for MI455X (gfx1250, wave32).
// Strategy: all spectral convs touch only 16x8 / 32x16 Fourier modes, so every
// FFT is done as a truncated direct DFT (tiny matmuls, memory-bound).  The two
// attention GEMMs (Q.K^T over i=4096 and P.V over s=128) run on WMMA
// v_wmma_f32_16x16x32_f16 with f32 accumulation.  One 128-entry twiddle table
// e^{-2pi i m/128} serves every transform (len 64 uses stride-2 indices).
// ---------------------------------------------------------------------------

typedef __attribute__((ext_vector_type(16))) _Float16 v16h;
typedef __attribute__((ext_vector_type(8)))  float    v8f;

#define PI_F 3.14159265358979323846f

// ------------------------------- tables ------------------------------------

__global__ void k_tw(float2* tw) {
  int m = threadIdx.x;
  if (m < 128) {
    float ang = -2.0f * PI_F * (float)m / 128.0f;
    tw[m] = make_float2(cosf(ang), sinf(ang));
  }
}

// Fused (rfft2-rows truncate -> ifft-rows len64) kernel matrix for the 2x
// Fourier resample:  KRES[yo][y] = (1/16384) * sum_{ky in {0..31,96..127}}
//   e^{-2pi i ky y/128} * e^{+2pi i kyo yo/64},  kyo = ky or ky-64.
__global__ void k_kres(const float2* __restrict__ tw, float2* __restrict__ kres) {
  int yo = blockIdx.x;    // 0..63
  int y  = threadIdx.x;   // 0..127
  float2 s = make_float2(0.f, 0.f);
  for (int j = 0; j < 32; ++j) {
    // top: ky=j -> out row j
    float2 a = tw[(j * y) & 127];
    float2 b = tw[(2 * j * yo) & 127];     // conj(b) => e^{+i}
    s.x += a.x * b.x + a.y * b.y;
    s.y += a.y * b.x - a.x * b.y;
    // bottom: ky=96+j -> out row 32+j
    int ky = 96 + j, ko = 32 + j;
    float2 c = tw[(ky * y) & 127];
    float2 d = tw[(2 * ko * yo) & 127];
    s.x += c.x * d.x + c.y * d.y;
    s.y += c.y * d.x - c.x * d.y;
  }
  s.x *= (1.0f / 16384.0f);
  s.y *= (1.0f / 16384.0f);
  kres[yo * 128 + y] = s;
}

// --------------------------- instance norm ---------------------------------

// out = IN(in (+ in2)) * g[inst%chmod] + b[inst%chmod] (+ addAfter)
__global__ __launch_bounds__(256)
void k_inorm(const float* __restrict__ in, const float* __restrict__ in2,
             float* __restrict__ out, const float* __restrict__ addAfter,
             const float* __restrict__ g, const float* __restrict__ bta,
             int chmod, int spatial) {
  __shared__ float sS[256], sQ[256];
  int inst = blockIdx.x, tid = threadIdx.x;
  size_t base = (size_t)inst * spatial;
  float s = 0.f, q = 0.f;
  for (int i = tid; i < spatial; i += 256) {
    float v = in[base + i];
    if (in2) v += in2[base + i];
    s += v; q += v * v;
  }
  sS[tid] = s; sQ[tid] = q; __syncthreads();
  for (int k = 128; k > 0; k >>= 1) {
    if (tid < k) { sS[tid] += sS[tid + k]; sQ[tid] += sQ[tid + k]; }
    __syncthreads();
  }
  float mu  = sS[0] / (float)spatial;
  float var = sQ[0] / (float)spatial - mu * mu;
  float rs  = rsqrtf(var + 1e-5f);
  float gm = g   ? g[inst % chmod]   : 1.f;
  float bb = bta ? bta[inst % chmod] : 0.f;
  for (int i = tid; i < spatial; i += 256) {
    float v = in[base + i];
    if (in2) v += in2[base + i];
    float y = (v - mu) * rs * gm + bb;
    if (addAfter) y += addAfter[base + i];
    out[base + i] = y;
  }
}

// ------------------------- forward DFT, stage 1 ----------------------------

// C1[row][kx] = sum_x in[row][x] * e^{-2pi i kx x/128}, rows of length 128.
__global__ __launch_bounds__(256)
void k_dft_col(const float* __restrict__ in, float2* __restrict__ out,
               const float2* __restrict__ tw, int nRows, int KX) {
  int idx = blockIdx.x * blockDim.x + threadIdx.x;
  int total = nRows * KX;
  if (idx >= total) return;
  int row = idx / KX, kx = idx - row * KX;
  const float* p = in + (size_t)row * 128;
  float re = 0.f, im = 0.f;
  for (int x = 0; x < 128; ++x) {
    float v = p[x];
    float2 t = tw[(kx * x) & 127];
    re += v * t.x;
    im += v * t.y;
  }
  out[idx] = make_float2(re, im);
}

// -------------------- Fourier 2x downsample (skip path) --------------------

__global__ __launch_bounds__(256)
void k_resample(const float2* __restrict__ c1, const float2* __restrict__ kres,
                const float2* __restrict__ tw, float* __restrict__ ds64) {
  __shared__ float2 Z[64 * 33];
  int tok = blockIdx.x, tid = threadIdx.x;
  const float2* base = c1 + (size_t)tok * 128 * 33;
  for (int e = tid; e < 64 * 33; e += 256) {
    int yo = e / 33, kx = e - yo * 33;
    float2 acc = make_float2(0.f, 0.f);
    const float2* kr = kres + yo * 128;
    for (int y = 0; y < 128; ++y) {
      float2 c = base[y * 33 + kx];
      float2 k = kr[y];
      acc.x += c.x * k.x - c.y * k.y;
      acc.y += c.x * k.y + c.y * k.x;
    }
    Z[e] = acc;
  }
  __syncthreads();
  for (int p = tid; p < 4096; p += 256) {
    int yo = p >> 6, xo = p & 63;
    float s = 0.f;
    for (int kx = 0; kx < 33; ++kx) {
      float2 z = Z[yo * 33 + kx];
      float2 t = tw[(2 * kx * xo) & 127];
      float re = z.x * t.x + z.y * t.y;  // Re(z * e^{+i theta})
      s += ((kx == 0) || (kx == 32) ? 1.f : 2.f) * re;
    }
    ds64[(size_t)tok * 4096 + p] = s;
  }
}

// ---------------------- KQV modes (16x8, per token) ------------------------

__global__ __launch_bounds__(256)
void k_kqv_modes(const float2* __restrict__ c1, const float2* __restrict__ tw,
                 float2* __restrict__ mk) {
  int idx = blockIdx.x * blockDim.x + threadIdx.x;
  if (idx >= 256 * 16 * 8) return;
  int kx = idx & 7, r = (idx >> 3) & 15, tok = idx >> 7;
  int ky = (r < 8) ? r : (112 + r);   // bottom rows 120..127
  const float2* base = c1 + (size_t)tok * 128 * 33 + kx;
  float2 acc = make_float2(0.f, 0.f);
  for (int y = 0; y < 128; ++y) {
    float2 c = base[y * 33];
    float2 t = tw[(ky * y) & 127];
    acc.x += c.x * t.x - c.y * t.y;
    acc.y += c.x * t.y + c.y * t.x;
  }
  mk[idx] = make_float2(acc.x * (1.f / 16384.f), acc.y * (1.f / 16384.f));
}

// ---------------- K / Q spectral conv + resampled skip -> f16 --------------

__global__ __launch_bounds__(256)
void k_kq_spec(const float2* __restrict__ mk, const float* __restrict__ ds64,
               const float2* __restrict__ tw,
               const float* __restrict__ wr, const float* __restrict__ wi,
               const float* __restrict__ bias, const float* __restrict__ sw,
               const float* __restrict__ sb, int s, _Float16* __restrict__ dst) {
  __shared__ float2 Sm[16 * 8];
  __shared__ float2 Zr[64 * 8];
  int tok = blockIdx.x, a = blockIdx.y, tid = threadIdx.x;
  int b = tok >> 7, tt = tok & 127;
  if (tid < 128) {
    int r = tid >> 3, kx = tid & 7;
    float2 m = mk[((size_t)tok * 16 + r) * 8 + kx];
    int widx = ((s * 8 + a) * 16 + r) * 8 + kx;
    float wrv = wr[widx], wiv = wi[widx];
    Sm[tid] = make_float2(m.x * wrv - m.y * wiv, m.x * wiv + m.y * wrv);
  }
  __syncthreads();
  for (int e = tid; e < 64 * 8; e += 256) {
    int yo = e >> 3, kx = e & 7;
    float2 acc = make_float2(0.f, 0.f);
    for (int r = 0; r < 16; ++r) {
      int kyo = (r < 8) ? r : (48 + r);       // bottom out rows 56..63
      float2 t = tw[(2 * kyo * yo) & 127];
      float2 m = Sm[r * 8 + kx];
      acc.x += m.x * t.x + m.y * t.y;         // m * conj(t)
      acc.y += m.y * t.x - m.x * t.y;
    }
    Zr[e] = acc;
  }
  __syncthreads();
  float bv = bias[s * 8 + a], swv = sw[s * 8 + a], sbv = sb[s * 8 + a];
  for (int p = tid; p < 4096; p += 256) {
    int yo = p >> 6, xo = p & 63;
    float acc = 0.f;
    for (int kx = 0; kx < 8; ++kx) {
      float2 z = Zr[yo * 8 + kx];
      float2 t = tw[(2 * kx * xo) & 127];
      acc += (kx == 0 ? 1.f : 2.f) * (z.x * t.x + z.y * t.y);
    }
    float skipv = swv * ds64[(size_t)tok * 4096 + p] + sbv;
    dst[(((size_t)(b * 8 + a)) * 128 + tt) * 4096 + p] = (_Float16)(acc + bv + skipv);
  }
}

// ------------------- V spectral conv + skip (full res) ---------------------

__global__ __launch_bounds__(256)
void k_v_spec(const float2* __restrict__ mk, const float* __restrict__ tokn,
              const float2* __restrict__ tw,
              const float* __restrict__ wr, const float* __restrict__ wi,
              const float* __restrict__ bias, const float* __restrict__ sw,
              const float* __restrict__ sb, _Float16* __restrict__ dst) {
  __shared__ float2 Sm[16 * 8];
  __shared__ float2 Zr[128 * 8];
  int tok = blockIdx.x, a = blockIdx.y, tid = threadIdx.x;
  int b = tok >> 7, tt = tok & 127;
  if (tid < 128) {
    int r = tid >> 3, kx = tid & 7;
    float2 m = mk[((size_t)tok * 16 + r) * 8 + kx];
    int widx = (a * 16 + r) * 8 + kx;
    float wrv = wr[widx], wiv = wi[widx];
    Sm[tid] = make_float2(m.x * wrv - m.y * wiv, m.x * wiv + m.y * wrv);
  }
  __syncthreads();
  for (int e = tid; e < 128 * 8; e += 256) {
    int yo = e >> 3, kx = e & 7;
    float2 acc = make_float2(0.f, 0.f);
    for (int r = 0; r < 16; ++r) {
      int kyo = (r < 8) ? r : (112 + r);     // bottom out rows 120..127
      float2 t = tw[(kyo * yo) & 127];
      float2 m = Sm[r * 8 + kx];
      acc.x += m.x * t.x + m.y * t.y;
      acc.y += m.y * t.x - m.x * t.y;
    }
    Zr[e] = acc;
  }
  __syncthreads();
  float bv = bias[a], swv = sw[a], sbv = sb[a];
  for (int p = tid; p < 16384; p += 256) {
    int yo = p >> 7, xo = p & 127;
    float acc = 0.f;
    for (int kx = 0; kx < 8; ++kx) {
      float2 z = Zr[yo * 8 + kx];
      float2 t = tw[(kx * xo) & 127];
      acc += (kx == 0 ? 1.f : 2.f) * (z.x * t.x + z.y * t.y);
    }
    float skipv = swv * tokn[(size_t)tok * 16384 + p] + sbv;
    dst[(((size_t)(b * 8 + a)) * 128 + tt) * 16384 + p] = (_Float16)(acc + bv + skipv);
  }
}

// -------------------------- attention: scores ------------------------------
// scores[ba][t][s] = (1/64) * sum_i Q[ba][t][i] * K[ba][s][i],  i = 4096.
// Block = (m-tile, ba); 8 waves, wave w owns n-tile w.  WMMA f16->f32.

__global__ __launch_bounds__(256)
void k_scores(const _Float16* __restrict__ Qh, const _Float16* __restrict__ Kh,
              float* __restrict__ sc) {
  __shared__ _Float16 Qs[16][32];
  __shared__ _Float16 Ks[128][32];
  int mt = blockIdx.x;   // 0..7
  int ba = blockIdx.y;   // 0..15
  int tid = threadIdx.x, lane = tid & 31, wv = tid >> 5;
  const _Float16* qbase = Qh + (size_t)ba * 128 * 4096 + (size_t)mt * 16 * 4096;
  const _Float16* kbase = Kh + (size_t)ba * 128 * 4096;
  v8f acc = {0.f, 0.f, 0.f, 0.f, 0.f, 0.f, 0.f, 0.f};
  for (int kc = 0; kc < 128; ++kc) {
    int k0 = kc * 32;
    { // Q tile: 16x32, 2 halves/thread
      int e = tid * 2;
      int r = e >> 5, c = e & 31;
      Qs[r][c]     = qbase[(size_t)r * 4096 + k0 + c];
      Qs[r][c + 1] = qbase[(size_t)r * 4096 + k0 + c + 1];
    }
    { // K tile: 128x32, 16 halves/thread
      int r = tid >> 1, c0l = (tid & 1) * 16;
      const _Float16* src = kbase + (size_t)r * 4096 + k0 + c0l;
      #pragma unroll
      for (int j = 0; j < 16; ++j) Ks[r][c0l + j] = src[j];
    }
    __syncthreads();
    v16h af, bf;
    int m = lane & 15, g = lane >> 4;
    #pragma unroll
    for (int v = 0; v < 8; ++v) {
      int karr = ((v < 4) ? 0 : 16) + g * 8 + 2 * (v & 3);   // A layout
      af[2 * v]     = Qs[m][karr];
      af[2 * v + 1] = Qs[m][karr + 1];
      int kb = ((v < 4) ? 0 : 16) + g * 8 + 2 * (v & 3);     // B layout
      bf[2 * v]     = Ks[wv * 16 + m][kb];
      bf[2 * v + 1] = Ks[wv * 16 + m][kb + 1];
    }
    acc = __builtin_amdgcn_wmma_f32_16x16x32_f16(false, af, false, bf,
                                                 (short)0, acc, false, false);
    __syncthreads();
  }
  int m = lane & 15, g = lane >> 4;
  #pragma unroll
  for (int r = 0; r < 8; ++r) {
    int row = mt * 16 + r + 8 * g;
    int col = wv * 16 + m;
    sc[((size_t)ba * 128 + row) * 128 + col] = acc[r] * (1.0f / 64.0f);
  }
}

// --------------------------- softmax over s --------------------------------

__global__ __launch_bounds__(128)
void k_softmax(const float* __restrict__ sc, _Float16* __restrict__ pr) {
  __shared__ float sh[128];
  int row = blockIdx.x, t = threadIdx.x;
  float v = sc[(size_t)row * 128 + t];
  sh[t] = v; __syncthreads();
  for (int s = 64; s > 0; s >>= 1) { if (t < s) sh[t] = fmaxf(sh[t], sh[t + s]); __syncthreads(); }
  float mx = sh[0]; __syncthreads();
  float e = expf(v - mx);
  sh[t] = e; __syncthreads();
  for (int s = 64; s > 0; s >>= 1) { if (t < s) sh[t] += sh[t + s]; __syncthreads(); }
  pr[(size_t)row * 128 + t] = (_Float16)(e / sh[0]);
}

// ---------------------------- attention: P.V -------------------------------
// att[(b*128+t)*8+a][pix] = sum_s P[ba][t][s] * V[ba][s][pix].  M=K=128, N=16384.

__global__ __launch_bounds__(256)
void k_attnv(const _Float16* __restrict__ Pr, const _Float16* __restrict__ Vh,
             float* __restrict__ att) {
  __shared__ _Float16 Ps[128][128];
  __shared__ _Float16 Vs[32][128];
  int cc = blockIdx.x;   // column chunk, 0..127
  int ba = blockIdx.y;   // 0..15
  int b = ba >> 3, a = ba & 7;
  int tid = threadIdx.x, lane = tid & 31, wv = tid >> 5;
  int c0 = cc * 128;
  { // probs 128x128, 64 halves/thread
    int r = tid >> 1, o = (tid & 1) * 64;
    const _Float16* src = Pr + ((size_t)ba * 128 + r) * 128 + o;
    for (int j = 0; j < 64; ++j) Ps[r][o + j] = src[j];
  }
  const _Float16* vbase = Vh + (size_t)ba * 128 * 16384;
  v8f zero = {0.f, 0.f, 0.f, 0.f, 0.f, 0.f, 0.f, 0.f};
  v8f acc[8];
  #pragma unroll
  for (int i = 0; i < 8; ++i) acc[i] = zero;
  for (int kc = 0; kc < 4; ++kc) {
    __syncthreads();
    { // V chunk 32x128, 16 halves/thread
      int r = tid >> 3, seg = (tid & 7) * 16;
      const _Float16* src = vbase + ((size_t)(kc * 32 + r)) * 16384 + c0 + seg;
      for (int j = 0; j < 16; ++j) Vs[r][seg + j] = src[j];
    }
    __syncthreads();
    int m = lane & 15, g = lane >> 4;
    v16h bf;
    #pragma unroll
    for (int v = 0; v < 8; ++v) {
      int kb = ((v < 4) ? 0 : 16) + g * 8 + 2 * (v & 3);
      bf[2 * v]     = Vs[kb][wv * 16 + m];
      bf[2 * v + 1] = Vs[kb + 1][wv * 16 + m];
    }
    for (int mt = 0; mt < 8; ++mt) {
      v16h af;
      #pragma unroll
      for (int v = 0; v < 8; ++v) {
        int karr = kc * 32 + ((v < 4) ? 0 : 16) + g * 8 + 2 * (v & 3);
        af[2 * v]     = Ps[mt * 16 + m][karr];
        af[2 * v + 1] = Ps[mt * 16 + m][karr + 1];
      }
      acc[mt] = __builtin_amdgcn_wmma_f32_16x16x32_f16(false, af, false, bf,
                                                       (short)0, acc[mt], false, false);
    }
  }
  int m = lane & 15, g = lane >> 4;
  for (int mt = 0; mt < 8; ++mt)
    for (int r = 0; r < 8; ++r) {
      int t = mt * 16 + r + 8 * g;
      int col = c0 + wv * 16 + m;
      att[(((size_t)(b * 128 + t)) * 8 + a) * 16384 + col] = acc[mt][r];
    }
}

// ----------------- projection modes (heads -> codim 1, 32x16) --------------

__global__ __launch_bounds__(256)
void k_proj_modes(const float2* __restrict__ c1p, const float2* __restrict__ tw,
                  const float* __restrict__ pwr, const float* __restrict__ pwi,
                  float2* __restrict__ mp) {
  int idx = blockIdx.x * blockDim.x + threadIdx.x;
  if (idx >= 256 * 32 * 16) return;
  int kx = idx & 15, r = (idx >> 4) & 31, tok = idx >> 9;
  int ky = (r < 16) ? r : (96 + r);    // bottom rows 112..127
  float2 acc = make_float2(0.f, 0.f);
  for (int a = 0; a < 8; ++a) {
    const float2* base = c1p + (((size_t)tok * 8 + a) * 128) * 16 + kx;
    float2 xa = make_float2(0.f, 0.f);
    for (int y = 0; y < 128; ++y) {
      float2 c = base[y * 16];
      float2 t = tw[(ky * y) & 127];
      xa.x += c.x * t.x - c.y * t.y;
      xa.y += c.x * t.y + c.y * t.x;
    }
    int widx = (a * 32 + r) * 16 + kx;
    float wrv = pwr[widx], wiv = pwi[widx];
    acc.x += xa.x * wrv - xa.y * wiv;
    acc.y += xa.x * wiv + xa.y * wrv;
  }
  mp[idx] = make_float2(acc.x * (1.f / 16384.f), acc.y * (1.f / 16384.f));
}

// --------------- projection irfft2 + bias + 1x1 head skip ------------------

__global__ __launch_bounds__(256)
void k_proj_out(const float2* __restrict__ mp, const float* __restrict__ att,
                const float2* __restrict__ tw, const float* __restrict__ psw,
                const float* __restrict__ psb, const float* __restrict__ pbias,
                float* __restrict__ out) {
  __shared__ float2 Mo[32 * 16];
  __shared__ float2 Zr[128 * 16];
  int tok = blockIdx.x, tid = threadIdx.x;
  for (int e = tid; e < 512; e += 256) Mo[e] = mp[(size_t)tok * 512 + e];
  __syncthreads();
  for (int e = tid; e < 2048; e += 256) {
    int y = e >> 4, kx = e & 15;
    float2 acc = make_float2(0.f, 0.f);
    for (int r = 0; r < 32; ++r) {
      int kyo = (r < 16) ? r : (96 + r);
      float2 t = tw[(kyo * y) & 127];
      float2 m = Mo[r * 16 + kx];
      acc.x += m.x * t.x + m.y * t.y;
      acc.y += m.y * t.x - m.x * t.y;
    }
    Zr[e] = acc;
  }
  __syncthreads();
  float swl[8];
  for (int a = 0; a < 8; ++a) swl[a] = psw[a];
  float bv = pbias[0] + psb[0];
  for (int p = tid; p < 16384; p += 256) {
    int y = p >> 7, x = p & 127;
    float acc = 0.f;
    for (int kx = 0; kx < 16; ++kx) {
      float2 z = Zr[y * 16 + kx];
      float2 t = tw[(kx * x) & 127];
      acc += (kx == 0 ? 1.f : 2.f) * (z.x * t.x + z.y * t.y);
    }
    float skip = 0.f;
    for (int a = 0; a < 8; ++a) skip += att[((size_t)tok * 8 + a) * 16384 + p] * swl[a];
    out[(size_t)tok * 16384 + p] = acc + bv + skip;
  }
}

// ------------------------- mixer modes (4->4, 32x16) -----------------------

__global__ __launch_bounds__(256)
void k_mix_modes(const float2* __restrict__ m1, const float2* __restrict__ tw,
                 const float* __restrict__ mwr, const float* __restrict__ mwi,
                 float2* __restrict__ mm) {
  int idx = blockIdx.x * blockDim.x + threadIdx.x;
  if (idx >= 64 * 4 * 32 * 16) return;
  int kx = idx & 15, r = (idx >> 4) & 31, o = (idx >> 9) & 3, smp = idx >> 11;
  int ky = (r < 16) ? r : (96 + r);
  float2 acc = make_float2(0.f, 0.f);
  for (int i = 0; i < 4; ++i) {
    const float2* base = m1 + (((size_t)smp * 4 + i) * 128) * 16 + kx;
    float2 xa = make_float2(0.f, 0.f);
    for (int y = 0; y < 128; ++y) {
      float2 c = base[y * 16];
      float2 t = tw[(ky * y) & 127];
      xa.x += c.x * t.x - c.y * t.y;
      xa.y += c.x * t.y + c.y * t.x;
    }
    int widx = ((i * 4 + o) * 32 + r) * 16 + kx;
    float wrv = mwr[widx], wiv = mwi[widx];
    acc.x += xa.x * wrv - xa.y * wiv;
    acc.y += xa.x * wiv + xa.y * wrv;
  }
  mm[idx] = make_float2(acc.x * (1.f / 16384.f), acc.y * (1.f / 16384.f));
}

// ------------------------- mixer irfft2 + bias -----------------------------

__global__ __launch_bounds__(256)
void k_irfft2_128(const float2* __restrict__ mm, const float2* __restrict__ tw,
                  const float* __restrict__ bias, float* __restrict__ out) {
  __shared__ float2 Mo[32 * 16];
  __shared__ float2 Zr[128 * 16];
  int inst = blockIdx.x, tid = threadIdx.x;
  for (int e = tid; e < 512; e += 256) Mo[e] = mm[(size_t)inst * 512 + e];
  __syncthreads();
  for (int e = tid; e < 2048; e += 256) {
    int y = e >> 4, kx = e & 15;
    float2 acc = make_float2(0.f, 0.f);
    for (int r = 0; r < 32; ++r) {
      int kyo = (r < 16) ? r : (96 + r);
      float2 t = tw[(kyo * y) & 127];
      float2 m = Mo[r * 16 + kx];
      acc.x += m.x * t.x + m.y * t.y;
      acc.y += m.y * t.x - m.x * t.y;
    }
    Zr[e] = acc;
  }
  __syncthreads();
  float bv = bias[inst & 3];
  for (int p = tid; p < 16384; p += 256) {
    int y = p >> 7, x = p & 127;
    float acc = 0.f;
    for (int kx = 0; kx < 16; ++kx) {
      float2 z = Zr[y * 16 + kx];
      float2 t = tw[(kx * x) & 127];
      acc += (kx == 0 ? 1.f : 2.f) * (z.x * t.x + z.y * t.y);
    }
    out[(size_t)inst * 16384 + p] = acc + bv;
  }
}

// ------------- mixer: IN(spec) + 1x1 skip (+ exact gelu) -------------------

__global__ __launch_bounds__(256)
void k_mix_post(const float* __restrict__ S, const float* __restrict__ hin,
                const float* __restrict__ sw, const float* __restrict__ sb,
                int do_gelu, float* __restrict__ out) {
  __shared__ float sS[256], sQ[256];
  int inst = blockIdx.x, tid = threadIdx.x;
  int smp = inst >> 2, o = inst & 3;
  const float* base = S + (size_t)inst * 16384;
  float s = 0.f, q = 0.f;
  for (int i = tid; i < 16384; i += 256) { float v = base[i]; s += v; q += v * v; }
  sS[tid] = s; sQ[tid] = q; __syncthreads();
  for (int k = 128; k > 0; k >>= 1) {
    if (tid < k) { sS[tid] += sS[tid + k]; sQ[tid] += sQ[tid + k]; }
    __syncthreads();
  }
  float mu  = sS[0] * (1.f / 16384.f);
  float var = sQ[0] * (1.f / 16384.f) - mu * mu;
  float rs  = rsqrtf(var + 1e-5f);
  float sw0 = sw[o * 4 + 0], sw1 = sw[o * 4 + 1], sw2 = sw[o * 4 + 2], sw3 = sw[o * 4 + 3];
  float sbv = sb[o];
  const float* h0b = hin + (size_t)smp * 4 * 16384;
  for (int i = tid; i < 16384; i += 256) {
    float v = (base[i] - mu) * rs;
    float skip = sbv + h0b[i] * sw0 + h0b[16384 + i] * sw1 +
                 h0b[2 * 16384 + i] * sw2 + h0b[3 * 16384 + i] * sw3;
    float y = v + skip;
    if (do_gelu) y = 0.5f * y * (1.0f + erff(y * 0.70710678118654752f));
    out[(size_t)inst * 16384 + i] = y;
  }
}

// ---------------------------------------------------------------------------

extern "C" void kernel_launch(void* const* d_in, const int* in_sizes, int n_in,
                              void* d_out, int out_size, void* d_ws, size_t ws_size,
                              hipStream_t stream) {
  (void)in_sizes; (void)n_in; (void)out_size; (void)ws_size;
  const float* x           = (const float*)d_in[0];
  const float* norm1_w     = (const float*)d_in[1];
  const float* norm1_b     = (const float*)d_in[2];
  const float* kqv_wr      = (const float*)d_in[3];
  const float* kqv_wi      = (const float*)d_in[4];
  const float* kqv_bias    = (const float*)d_in[5];
  const float* kqv_sw      = (const float*)d_in[6];
  const float* kqv_sb      = (const float*)d_in[7];
  const float* p_wr        = (const float*)d_in[8];
  const float* p_wi        = (const float*)d_in[9];
  const float* p_bias      = (const float*)d_in[10];
  const float* p_sw        = (const float*)d_in[11];
  const float* p_sb        = (const float*)d_in[12];
  const float* attn_norm_w = (const float*)d_in[13];
  const float* attn_norm_b = (const float*)d_in[14];
  const float* norm2_w     = (const float*)d_in[15];
  const float* norm2_b     = (const float*)d_in[16];
  const float* mix_wr      = (const float*)d_in[17];
  const float* mix_wi      = (const float*)d_in[18];
  const float* mix_bias    = (const float*)d_in[19];
  const float* mix_sw      = (const float*)d_in[20];
  const float* mix_sb      = (const float*)d_in[21];
  const float* mon_w       = (const float*)d_in[22];
  const float* mon_b       = (const float*)d_in[23];
  float* out = (float*)d_out;

  char* w = (char*)d_ws;
  size_t off = 0;
  auto alloc = [&](size_t bytes) -> void* {
    void* p = w + off;
    off += (bytes + 255) & ~(size_t)255;
    return p;
  };

  float2*   TW   = (float2*)  alloc(128 * sizeof(float2));
  float2*   KRES = (float2*)  alloc(64 * 128 * sizeof(float2));
  float*    TOKN = (float*)   alloc((size_t)256 * 16384 * 4);
  float2*   DS1  = (float2*)  alloc((size_t)32768 * 33 * 8);
  float*    DS64 = (float*)   alloc((size_t)256 * 4096 * 4);
  float2*   MK   = (float2*)  alloc((size_t)32768 * 8);
  _Float16* KH   = (_Float16*)alloc((size_t)16 * 128 * 4096 * 2);
  _Float16* QH   = (_Float16*)alloc((size_t)16 * 128 * 4096 * 2);
  _Float16* VH   = (_Float16*)alloc((size_t)16 * 128 * 16384 * 2);
  float*    SC   = (float*)   alloc((size_t)16 * 128 * 128 * 4);
  _Float16* PRH  = (_Float16*)alloc((size_t)16 * 128 * 128 * 2);
  float*    ATT  = (float*)   alloc((size_t)256 * 8 * 16384 * 4);
  float2*   C1P  = (float2*)  alloc((size_t)262144 * 16 * 8);
  float2*   MP   = (float2*)  alloc((size_t)256 * 512 * 8);
  float*    S    = (float*)   alloc((size_t)256 * 16384 * 4);
  float*    A2   = (float*)   alloc((size_t)256 * 16384 * 4);
  float*    H0   = (float*)   alloc((size_t)256 * 16384 * 4);
  float2*   MX1  = (float2*)  alloc((size_t)32768 * 16 * 8);
  float2*   MMX  = (float2*)  alloc((size_t)256 * 512 * 8);
  float*    H1   = (float*)   alloc((size_t)256 * 16384 * 4);
  float*    H2   = (float*)   alloc((size_t)256 * 16384 * 4);

  // ---- tables ----
  k_tw  <<<1, 128, 0, stream>>>(TW);
  k_kres<<<64, 128, 0, stream>>>(TW, KRES);

  // ---- norm1 on per-channel tokens ----
  k_inorm<<<256, 256, 0, stream>>>(x, nullptr, TOKN, nullptr, norm1_w, norm1_b, 1, 16384);

  // ---- stage-1 column DFT of tokens (33 cols covers resample + kqv modes) ----
  k_dft_col<<<(32768 * 33 + 255) / 256, 256, 0, stream>>>(TOKN, DS1, TW, 32768, 33);

  // ---- Fourier 2x downsample of tokens (skip for K/Q) ----
  k_resample<<<256, 256, 0, stream>>>(DS1, KRES, TW, DS64);

  // ---- KQV low modes (16x8) ----
  k_kqv_modes<<<(32768 + 255) / 256, 256, 0, stream>>>(DS1, TW, MK);

  // ---- K, Q, V spectral conv + skip, write head-major f16 ----
  k_kq_spec<<<dim3(256, 8), 256, 0, stream>>>(MK, DS64, TW, kqv_wr, kqv_wi,
                                              kqv_bias, kqv_sw, kqv_sb, 0, KH);
  k_kq_spec<<<dim3(256, 8), 256, 0, stream>>>(MK, DS64, TW, kqv_wr, kqv_wi,
                                              kqv_bias, kqv_sw, kqv_sb, 1, QH);
  k_v_spec <<<dim3(256, 8), 256, 0, stream>>>(MK, TOKN, TW,
                                              kqv_wr + 2 * 1024, kqv_wi + 2 * 1024,
                                              kqv_bias + 16, kqv_sw + 16, kqv_sb + 16, VH);

  // ---- attention (WMMA) ----
  k_scores <<<dim3(8, 16), 256, 0, stream>>>(QH, KH, SC);
  k_softmax<<<2048, 128, 0, stream>>>(SC, PRH);
  k_attnv  <<<dim3(128, 16), 256, 0, stream>>>(PRH, VH, ATT);

  // ---- projection FNO layer (heads -> codim 1) ----
  k_dft_col  <<<(262144 * 16 + 255) / 256, 256, 0, stream>>>(ATT, C1P, TW, 262144, 16);
  k_proj_modes<<<(131072 + 255) / 256, 256, 0, stream>>>(C1P, TW, p_wr, p_wi, MP);
  k_proj_out <<<256, 256, 0, stream>>>(MP, ATT, TW, p_sw, p_sb, p_bias, S);

  // ---- attn_norm(proj + tokens) -> A2 ----
  k_inorm<<<256, 256, 0, stream>>>(S, x, A2, nullptr, attn_norm_w, attn_norm_b, 1, 16384);

  // ---- mixer: norm2 then two FNO layers ----
  k_inorm<<<256, 256, 0, stream>>>(A2, nullptr, H0, nullptr, norm2_w, norm2_b, 4, 16384);

  // layer 0
  k_dft_col   <<<(32768 * 16 + 255) / 256, 256, 0, stream>>>(H0, MX1, TW, 32768, 16);
  k_mix_modes <<<(131072 + 255) / 256, 256, 0, stream>>>(MX1, TW, mix_wr, mix_wi, MMX);
  k_irfft2_128<<<256, 256, 0, stream>>>(MMX, TW, mix_bias, S);
  k_mix_post  <<<256, 256, 0, stream>>>(S, H0, mix_sw, mix_sb, 1, H1);

  // layer 1
  k_dft_col   <<<(32768 * 16 + 255) / 256, 256, 0, stream>>>(H1, MX1, TW, 32768, 16);
  k_mix_modes <<<(131072 + 255) / 256, 256, 0, stream>>>(MX1, TW, mix_wr + 2048, mix_wi + 2048, MMX);
  k_irfft2_128<<<256, 256, 0, stream>>>(MMX, TW, mix_bias + 4, S);
  k_mix_post  <<<256, 256, 0, stream>>>(S, H1, mix_sw + 16, mix_sb + 4, 0, H2);

  // ---- final: IN(h2, mon) + attn ----
  k_inorm<<<256, 256, 0, stream>>>(H2, nullptr, out, A2, mon_w, mon_b, 4, 16384);
}
